// MultiHeadSeqAttention_82789789597729
// MI455X (gfx1250) — compile-verified, matrix operand order checked by reading, not run on hardware
//
#include <hip/hip_runtime.h>

// MI455X / gfx1250, wave32. bf16 WMMA (f32 accum) pipeline.
// Pipeline (softmax in the reference is dead code; (QK^T)V == Q(K^T V)):
//   cvt:  fp32 -> bf16 for h, h_cache, Wq, Wk, Wv, Wo
//   proj: Q = h Wq^T (row-major), Kt/Vt = (h_cache Wk/v^T)^T  (transposed store)
//   g:    G[b,hd] = Kt_head @ Vt_head^T  (= K^T V), split-K f32 atomics
//   w2:   W2t[b] = (G Wo_head^T)^T  per head, transposed store
//   out:  out = Q @ W2t^T  (fp32 store)
// Big GEMMs use 64x64-per-wave register blocking: 16 b128 loads : 16 WMMAs.

typedef __bf16 bf16_t;
typedef __attribute__((ext_vector_type(4)))  __bf16 v4bf;
typedef __attribute__((ext_vector_type(16))) __bf16 v16bf;
typedef __attribute__((ext_vector_type(8)))  float  v8f;

#define BATCH  2
#define SEQ    2048
#define HDIM   1024
#define HEADS  16
#define DHEAD  64
#define KSPLIT 16

// ---------------- elementwise helpers ----------------

__global__ void cvt_f32_to_bf16(const float* __restrict__ s, bf16_t* __restrict__ d, int n4) {
  int i = blockIdx.x * blockDim.x + threadIdx.x;
  if (i < n4) {
    float4 v = ((const float4*)s)[i];
    v4bf o;
    o[0] = (bf16_t)v.x; o[1] = (bf16_t)v.y; o[2] = (bf16_t)v.z; o[3] = (bf16_t)v.w;
    *(v4bf*)(d + 4 * (size_t)i) = o;
  }
}

__global__ void zero_f32(float* __restrict__ p, int n) {
  int i = blockIdx.x * blockDim.x + threadIdx.x;
  if (i < n) p[i] = 0.0f;
}

// ---------------- WMMA NT-GEMM core (templated register blocking) ----------------

__device__ __forceinline__ v16bf frag_nt(const bf16_t* __restrict__ p, int ld, int k) {
  const int lane = threadIdx.x & 31;
  // lane&15 -> row within 16-row panel; lane>>4 selects K 0..15 vs 16..31 half
  return *(const v16bf*)(p + (size_t)(lane & 15) * ld + k + ((lane >> 4) << 4));
}

__device__ __forceinline__ v8f wmma_bf16(v16bf a, v16bf b, v8f c) {
  return __builtin_amdgcn_wmma_f32_16x16x32_bf16(false, a, false, b, (short)0, c,
                                                 false, false);
}

// Wave computes a (MW*16) x (NW*16) tile; K in steps of 32.
template <int MW, int NW>
__device__ __forceinline__ void gemm_nt_acc(const bf16_t* __restrict__ A, int lda,
                                            const bf16_t* __restrict__ B, int ldb,
                                            int K, v8f acc[MW][NW]) {
  const int lane = threadIdx.x & 31;
  for (int k = 0; k < K; k += 32) {
    v16bf af[MW];
    v16bf bfv[NW];
#pragma unroll
    for (int i = 0; i < MW; ++i) af[i] = frag_nt(A + (size_t)(16 * i) * lda, lda, k);
#pragma unroll
    for (int j = 0; j < NW; ++j) bfv[j] = frag_nt(B + (size_t)(16 * j) * ldb, ldb, k);
    // speculative prefetch of the K-stream (OOB prefetches silently dropped)
    __builtin_prefetch(A + (size_t)(lane & 15) * lda + k + 512, 0, 3);
    __builtin_prefetch(B + (size_t)(lane & 15) * ldb + k + 512, 0, 3);
#pragma unroll
    for (int i = 0; i < MW; ++i)
#pragma unroll
      for (int j = 0; j < NW; ++j)
        acc[i][j] = wmma_bf16(af[i], bfv[j], acc[i][j]);
  }
}

template <int MW, int NW>
__device__ __forceinline__ void zero_acc(v8f acc[MW][NW]) {
  const v8f vz = {};
#pragma unroll
  for (int i = 0; i < MW; ++i)
#pragma unroll
    for (int j = 0; j < NW; ++j) acc[i][j] = vz;
}

// ---------------- tile stores (C/D layout: vgpr r -> row r+8*(lane>>4), col lane&15)

__device__ __forceinline__ void store_tile_bf16(bf16_t* __restrict__ C, int ldc, v8f c) {
  const int lane = threadIdx.x & 31;
  const int col = lane & 15, rb = (lane >> 4) * 8;
#pragma unroll
  for (int r = 0; r < 8; ++r) C[(size_t)(rb + r) * ldc + col] = (bf16_t)c[r];
}

__device__ __forceinline__ void store_tile_bf16_T(bf16_t* __restrict__ Ct, int ldt, v8f c) {
  const int lane = threadIdx.x & 31;
  const int col = lane & 15, rb = (lane >> 4) * 8;  // col = original n, rb+r = original m
#pragma unroll
  for (int r = 0; r < 8; ++r) Ct[(size_t)col * ldt + rb + r] = (bf16_t)c[r];
}

__device__ __forceinline__ void store_tile_f32(float* __restrict__ C, int ldc, v8f c) {
  const int lane = threadIdx.x & 31;
  const int col = lane & 15, rb = (lane >> 4) * 8;
#pragma unroll
  for (int r = 0; r < 8; ++r) C[(size_t)(rb + r) * ldc + col] = c[r];
}

__device__ __forceinline__ void atomic_add_tile(float* __restrict__ C, int ldc, v8f c) {
  const int lane = threadIdx.x & 31;
  const int col = lane & 15, rb = (lane >> 4) * 8;
#pragma unroll
  for (int r = 0; r < 8; ++r) atomicAdd(&C[(size_t)(rb + r) * ldc + col], c[r]);
}

template <int MW, int NW>
__device__ __forceinline__ void store_acc_bf16(bf16_t* __restrict__ C, int ldc,
                                               v8f acc[MW][NW]) {
#pragma unroll
  for (int i = 0; i < MW; ++i)
#pragma unroll
    for (int j = 0; j < NW; ++j)
      store_tile_bf16(C + (size_t)(16 * i) * ldc + 16 * j, ldc, acc[i][j]);
}

// original (m,n) tile (i,j) lands at T[col=n][row=m]: +16j rows, +16i cols in T
template <int MW, int NW>
__device__ __forceinline__ void store_acc_bf16_T(bf16_t* __restrict__ T, int ldt,
                                                 v8f acc[MW][NW]) {
#pragma unroll
  for (int i = 0; i < MW; ++i)
#pragma unroll
    for (int j = 0; j < NW; ++j)
      store_tile_bf16_T(T + (size_t)(16 * j) * ldt + 16 * i, ldt, acc[i][j]);
}

template <int MW, int NW>
__device__ __forceinline__ void store_acc_f32(float* __restrict__ C, int ldc,
                                              v8f acc[MW][NW]) {
#pragma unroll
  for (int i = 0; i < MW; ++i)
#pragma unroll
    for (int j = 0; j < NW; ++j)
      store_tile_f32(C + (size_t)(16 * i) * ldc + 16 * j, ldc, acc[i][j]);
}

// ---------------- kernels ----------------

// z=0: Q = h @ Wq^T (row-major). z=1/2: Kt/Vt = (h_cache @ Wk/v^T)^T -> [B][H][SEQ]
// Block = 4 waves (2x2), wave tile 64x64 -> block tile 128x128.
__global__ __launch_bounds__(128) void proj_kernel(
    const bf16_t* __restrict__ h, const bf16_t* __restrict__ hc,
    const bf16_t* __restrict__ Wq, const bf16_t* __restrict__ Wk,
    const bf16_t* __restrict__ Wv,
    bf16_t* __restrict__ Q, bf16_t* __restrict__ Kt, bf16_t* __restrict__ Vt) {
  const int w  = threadIdx.x >> 5;
  const int m0 = blockIdx.x * 128 + (w >> 1) * 64;  // row over B*SEQ = 4096
  const int n0 = blockIdx.y * 128 + (w & 1) * 64;   // output feature
  const int z  = blockIdx.z;
  const bf16_t* A = (z == 0 ? h : hc) + (size_t)m0 * HDIM;
  const bf16_t* W = (z == 0 ? Wq : (z == 1 ? Wk : Wv));
  const bf16_t* B = W + (size_t)n0 * HDIM;
  v8f acc[4][4];
  zero_acc<4, 4>(acc);
  gemm_nt_acc<4, 4>(A, HDIM, B, HDIM, HDIM, acc);
  if (z == 0) {
    store_acc_bf16<4, 4>(Q + (size_t)m0 * HDIM + n0, HDIM, acc);
  } else {
    const int b  = m0 >> 11;  // batch (SEQ=2048 rows each; 2048 % 128 == 0)
    const int l0 = m0 & (SEQ - 1);
    bf16_t* T = (z == 1 ? Kt : Vt) + (size_t)b * HDIM * SEQ + (size_t)n0 * SEQ + l0;
    store_acc_bf16_T<4, 4>(T, SEQ, acc);
  }
}

// G[z=b*HEADS+hd] = Kt_head @ Vt_head^T, split-K over SEQ with f32 atomics
__global__ __launch_bounds__(128) void g_kernel(
    const bf16_t* __restrict__ Kt, const bf16_t* __restrict__ Vt,
    float* __restrict__ Gf) {
  const int z  = blockIdx.z;
  const int w  = threadIdx.x >> 5;
  const int m0 = (w >> 1) * 32;  // d1
  const int n0 = (w & 1) * 32;   // d2
  const int kb = blockIdx.x * (SEQ / KSPLIT);
  const size_t hoff = (size_t)z * DHEAD * SEQ;  // == b*HDIM*SEQ + hd*DHEAD*SEQ
  const bf16_t* A = Kt + hoff + (size_t)m0 * SEQ + kb;
  const bf16_t* B = Vt + hoff + (size_t)n0 * SEQ + kb;
  v8f acc[2][2];
  zero_acc<2, 2>(acc);
  gemm_nt_acc<2, 2>(A, SEQ, B, SEQ, SEQ / KSPLIT, acc);
  float* G = Gf + (size_t)z * DHEAD * DHEAD + (size_t)m0 * DHEAD + n0;
#pragma unroll
  for (int i = 0; i < 2; ++i)
#pragma unroll
    for (int j = 0; j < 2; ++j)
      atomic_add_tile(G + (size_t)(16 * i) * DHEAD + 16 * j, DHEAD, acc[i][j]);
}

// W2t[b][o][hd*64+d] = sum_d2 G[b,hd][d][d2] * Wo[o][hd*64+d2]
__global__ __launch_bounds__(128) void w2_kernel(
    const bf16_t* __restrict__ G, const bf16_t* __restrict__ Wo,
    bf16_t* __restrict__ W2t) {
  const int z  = blockIdx.z;
  const int b  = z >> 4, hd = z & 15;
  const int w  = threadIdx.x >> 5;
  const int m0 = (w >> 1) * 32;                   // d (0..63)
  const int n0 = blockIdx.y * 64 + (w & 1) * 32;  // o
  const bf16_t* A = G + (size_t)z * DHEAD * DHEAD + (size_t)m0 * DHEAD;
  const bf16_t* B = Wo + (size_t)n0 * HDIM + hd * DHEAD;
  v8f acc[2][2];
  zero_acc<2, 2>(acc);
  gemm_nt_acc<2, 2>(A, DHEAD, B, HDIM, DHEAD, acc);
  bf16_t* T = W2t + (size_t)b * HDIM * HDIM + (size_t)n0 * HDIM + hd * DHEAD + m0;
  store_acc_bf16_T<2, 2>(T, HDIM, acc);
}

// out[b] = Q[b] @ W2t[b]^T  (fp32 output). 64x64 per wave, 128x128 per block.
__global__ __launch_bounds__(128) void out_kernel(
    const bf16_t* __restrict__ Q, const bf16_t* __restrict__ W2t,
    float* __restrict__ out) {
  const int b  = blockIdx.z;
  const int w  = threadIdx.x >> 5;
  const int m0 = blockIdx.x * 128 + (w >> 1) * 64;
  const int n0 = blockIdx.y * 128 + (w & 1) * 64;
  const bf16_t* A = Q + (size_t)b * SEQ * HDIM + (size_t)m0 * HDIM;
  const bf16_t* B = W2t + (size_t)b * HDIM * HDIM + (size_t)n0 * HDIM;
  v8f acc[4][4];
  zero_acc<4, 4>(acc);
  gemm_nt_acc<4, 4>(A, HDIM, B, HDIM, HDIM, acc);
  store_acc_f32<4, 4>(out + (size_t)b * SEQ * HDIM + (size_t)m0 * HDIM + n0, HDIM, acc);
}

// ---------------- launch ----------------

extern "C" void kernel_launch(void* const* d_in, const int* in_sizes, int n_in,
                              void* d_out, int out_size, void* d_ws, size_t ws_size,
                              hipStream_t stream) {
  const float* h  = (const float*)d_in[0];
  const float* hc = (const float*)d_in[1];
  // d_in[2] = key_pe: dead code in the reference (softmax result is unused)
  const float* Wq = (const float*)d_in[3];
  const float* Wk = (const float*)d_in[4];
  const float* Wv = (const float*)d_in[5];
  const float* Wo = (const float*)d_in[6];

  char* ws = (char*)d_ws;
  size_t off = 0;
  auto carve = [&](size_t bytes) -> void* {
    void* p = ws + off;
    off += (bytes + 255) & ~(size_t)255;
    return p;
  };
  const size_t NH = (size_t)BATCH * SEQ * HDIM;  // 4 Mi elements
  const size_t WW = (size_t)HDIM * HDIM;         // 1 Mi elements
  const size_t GG = (size_t)BATCH * HEADS * DHEAD * DHEAD;

  bf16_t* h_bf  = (bf16_t*)carve(NH * 2);
  bf16_t* hc_bf = (bf16_t*)carve(NH * 2);
  bf16_t* Wq_bf = (bf16_t*)carve(WW * 2);
  bf16_t* Wk_bf = (bf16_t*)carve(WW * 2);
  bf16_t* Wv_bf = (bf16_t*)carve(WW * 2);
  bf16_t* Wo_bf = (bf16_t*)carve(WW * 2);
  bf16_t* Q     = (bf16_t*)carve(NH * 2);
  bf16_t* Kt    = (bf16_t*)carve(NH * 2);
  bf16_t* Vt    = (bf16_t*)carve(NH * 2);
  float*  Gf    = (float*)carve(GG * 4);
  bf16_t* Gbf   = (bf16_t*)carve(GG * 2);
  bf16_t* W2t   = (bf16_t*)carve((size_t)BATCH * HDIM * HDIM * 2);

  const dim3 blk(128);

  cvt_f32_to_bf16<<<(unsigned)(NH / 4 / 256), 256, 0, stream>>>(h,  h_bf,  (int)(NH / 4));
  cvt_f32_to_bf16<<<(unsigned)(NH / 4 / 256), 256, 0, stream>>>(hc, hc_bf, (int)(NH / 4));
  cvt_f32_to_bf16<<<(unsigned)(WW / 4 / 256), 256, 0, stream>>>(Wq, Wq_bf, (int)(WW / 4));
  cvt_f32_to_bf16<<<(unsigned)(WW / 4 / 256), 256, 0, stream>>>(Wk, Wk_bf, (int)(WW / 4));
  cvt_f32_to_bf16<<<(unsigned)(WW / 4 / 256), 256, 0, stream>>>(Wv, Wv_bf, (int)(WW / 4));
  cvt_f32_to_bf16<<<(unsigned)(WW / 4 / 256), 256, 0, stream>>>(Wo, Wo_bf, (int)(WW / 4));

  zero_f32<<<(unsigned)(GG / 256), 256, 0, stream>>>(Gf, (int)GG);

  proj_kernel<<<dim3(BATCH * SEQ / 128, HDIM / 128, 3), blk, 0, stream>>>(
      h_bf, hc_bf, Wq_bf, Wk_bf, Wv_bf, Q, Kt, Vt);

  g_kernel<<<dim3(KSPLIT, 1, BATCH * HEADS), blk, 0, stream>>>(Kt, Vt, Gf);

  cvt_f32_to_bf16<<<(unsigned)(GG / 4 / 256), 256, 0, stream>>>(Gf, Gbf, (int)(GG / 4));

  w2_kernel<<<dim3(1, HDIM / 64, BATCH * HEADS), blk, 0, stream>>>(Gbf, Wo_bf, W2t);

  out_kernel<<<dim3(SEQ / 128, HDIM / 128, BATCH), blk, 0, stream>>>(Q, W2t, (float*)d_out);
}
